// BaseRuleLearner_54614804136586
// MI455X (gfx1250) — compile-verified, matrix operand order checked by reading, not run on hardware
//
#include <hip/hip_runtime.h>
#include <hip/hip_bf16.h>
#include <math.h>

typedef __attribute__((ext_vector_type(16))) _Float16 v16h;
typedef __attribute__((ext_vector_type(8)))  float    v8f;

// Problem sizes (static in the reference)
// I=8, IL=4, BL=25, P1=64, P2=16, B=64
// ws layout (floats): c1[2048] | c2[1536] | t1[64*800]
#define WS_C1 0
#define WS_C2 2048
#define WS_T1 (2048 + 1536)

// Raw hardware transcendentals (no denorm/inf guard sequences).
// Safe here: log2 argument is always in (0.7, 1]; exp2 argument is finite.
__device__ __forceinline__ float fast_log2(float x) { return __builtin_amdgcn_logf(x); }
__device__ __forceinline__ float fast_exp2(float x) { return __builtin_amdgcn_exp2f(x); }

// ---------------------------------------------------------------------------
// Kernel 1: inv conds = sigmoid(p) * feats  (tiny elementwise)
// ---------------------------------------------------------------------------
__global__ __launch_bounds__(256) void k_conds(const float* __restrict__ iuf,
                                               const float* __restrict__ ibf,
                                               const float* __restrict__ iup,
                                               const float* __restrict__ ibp,
                                               float* __restrict__ ws) {
    int x = blockIdx.x * blockDim.x + threadIdx.x;
    // sigmoid(t) = 1 / (1 + 2^(-t * log2(e)))
    const float kLog2e = 1.4426950408889634f;
    if (x < 2048) {
        float s = 1.0f / (1.0f + fast_exp2(-iup[x] * kLog2e));
        ws[WS_C1 + x] = s * iuf[x];
    } else if (x < 2048 + 1536) {
        int y = x - 2048;
        float s = 1.0f / (1.0f + fast_exp2(-ibp[y] * kLog2e));
        ws[WS_C2 + y] = s * ibf[y];
    }
}

// ---------------------------------------------------------------------------
// Kernel 2: t1[b,q,o] = prod_p (1 - c1[q,p]*(1-u[b,o,p]))
// Computed as exp2( sum_p log2(...) ), with the 64-wide p-sum done on the
// matrix pipe: two chained v_wmma_f32_16x16x32_f16 with B = ones.
// A-tile layout per CDNA5 ISA (16-bit A 16x32):
//   lanes 0-15  (M=lane):    elements 0..7 -> K=0..7,  8..15 -> K=16..23
//   lanes 16-31 (M=lane-16): elements 0..7 -> K=8..15, 8..15 -> K=24..31
// C/D layout: VGPR r holds (M=r, N=lane) for lanes 0-15, (M=r+8) for 16-31.
// One block per batch element b; 8 waves cover 50 16-output tiles (800 = 32q x 25o).
// ---------------------------------------------------------------------------
__global__ __launch_bounds__(256) void k_t1(const float* __restrict__ uf,
                                            const float* __restrict__ bfeat,
                                            float* __restrict__ ws) {
    __shared__ float c1s[2048];       // all inv unary conds (8KB)
    __shared__ float us[25 * 64];     // unary feats for this b (6.4KB)
    const int b   = blockIdx.x;
    const int tid = threadIdx.x;

    // Warm L2 with binary_feats for kernel 3 (global_prefetch_b8), overlapped
    // with this kernel's compute. 64*25*24*16 floats = 19200 cachelines.
    {
        const int lines = (64 * 25 * 24 * 16) / 32;
        for (int x = b * 256 + tid; x < lines; x += 64 * 256)
            __builtin_prefetch(bfeat + x * 32, 0, 1);
    }

    for (int x = tid; x < 2048; x += 256) c1s[x] = ws[WS_C1 + x];
    for (int x = tid; x < 1600; x += 256) us[x] = uf[b * 1600 + x];
    __syncthreads();

    const int wave = tid >> 5;
    const int lane = tid & 31;
    const int mm   = lane & 15;
    const int hi   = (lane >> 4) & 1;

    v16h bones;
#pragma unroll
    for (int e = 0; e < 16; ++e) bones[e] = (_Float16)1.0f;

    float* t1w = ws + WS_T1 + b * 800;

    for (int g = wave; g < 50; g += 8) {
        const int idx = 16 * g + mm;          // output index for this lane's A-row
        const int q   = idx / 25;             // (i*4+v)
        const int o   = idx % 25;
        const float* cq = &c1s[q * 64];
        const float* uo = &us[o * 64];

        v8f acc = {};
#pragma unroll
        for (int half = 0; half < 2; ++half) {
            const int base = half * 32;
            v16h a;
#pragma unroll
            for (int e = 0; e < 16; ++e) {
                const int p = base + (e < 8 ? e : e + 8) + (hi ? 8 : 0);
                const float c = cq[p];
                const float u = uo[p];
                const float x = fmaf(c, u, 1.0f - c);   // 1 - c*(1-u) in (0.7, 1]
                a[e] = (_Float16)fast_log2(x);          // bare v_log_f32
            }
            // D = A x ones + C : per-row sum over 32 p-values, f32 accumulate
            acc = __builtin_amdgcn_wmma_f32_16x16x32_f16(
                false, a, false, bones, (short)0, acc, false, false);
        }
        // Row-sums are replicated across columns; lane 0 holds M=0..7,
        // lane 16 holds M=8..15 (N=0 column).
        if (lane == 0) {
#pragma unroll
            for (int r = 0; r < 8; ++r) t1w[16 * g + r] = fast_exp2(acc[r]);
        } else if (lane == 16) {
#pragma unroll
            for (int r = 0; r < 8; ++r) t1w[16 * g + 8 + r] = fast_exp2(acc[r]);
        }
    }
}

// ---------------------------------------------------------------------------
// Kernel 3: fused t2 / bsat / uni_sets. One wave per (b,o); lane = i*4+v.
// uni[b,i,v,o] = t1[b,i,v,o] * prod_j (1 - prod_k (1 - t1[b,i,vj,ok]*t2))
// t2 = prod_p (1 - c2[i,v,j,p]*(1-bf[b,o,k,p]))  (16-wide fmaf chain in VALU,
// lowered to v_pk_fma_f32). All reused operands staged in LDS.
// ---------------------------------------------------------------------------
__global__ __launch_bounds__(32) void k_uni(const float* __restrict__ bfeat,
                                            const float* __restrict__ ws,
                                            float* __restrict__ out) {
    __shared__ float c2s[1536];
    __shared__ float bfs[24 * 16];
    __shared__ float t1s[800];
    const int blk = blockIdx.x;      // b*25 + o
    const int b   = blk / 25;
    const int o   = blk % 25;
    const int tid = threadIdx.x;     // 0..31 == i*4+v

    for (int x = tid; x < 1536; x += 32) c2s[x] = ws[WS_C2 + x];
    for (int x = tid; x < 384; x += 32)  bfs[x] = bfeat[(b * 25 + o) * 384 + x];
    for (int x = tid; x < 800; x += 32)  t1s[x] = ws[WS_T1 + b * 800 + x];
    __syncthreads();

    const int i = tid >> 2;
    const int v = tid & 3;
    const float t1self = t1s[tid * 25 + o];

    float acc = 1.0f;
#pragma unroll
    for (int j = 0; j < 3; ++j) {
        const int vj = j + (j >= v ? 1 : 0);      // vothers[v][j]
        float cr[16], mc[16];
        const float* c2r = &c2s[(tid * 3 + j) * 16];
#pragma unroll
        for (int p = 0; p < 16; ++p) { cr[p] = c2r[p]; mc[p] = 1.0f - cr[p]; }
        const float* t1g = &t1s[(i * 4 + vj) * 25];
        float prodk = 1.0f;
#pragma unroll 4
        for (int k = 0; k < 24; ++k) {
            const int ok = k + (k >= o ? 1 : 0);  // oothers[o][k]
            const float* bfk = &bfs[k * 16];
            float w = 1.0f;
#pragma unroll
            for (int p = 0; p < 16; ++p)
                w *= fmaf(cr[p], bfk[p], mc[p]);  // 1 - c*(1-f) = (1-c)+c*f
            const float x = t1g[ok] * w;
            prodk *= (1.0f - x);
        }
        acc *= (1.0f - prodk);
    }
    out[((b * 8 + i) * 4 + v) * 25 + o] = t1self * acc;
}

// ---------------------------------------------------------------------------
// Kernel 4: inv_uni = prod_v probsum_o(uni); inv_select = softmax(scale*inv_uni)
// One wave per b; lanes 0..7 handle the 8 invariants.
// ---------------------------------------------------------------------------
__global__ __launch_bounds__(32) void k_select(const float* __restrict__ uni,
                                               float* __restrict__ out) {
    __shared__ float sv[8];
    __shared__ float red[2];
    const int b    = blockIdx.x;
    const int lane = threadIdx.x;
    const float kScale = 8.8526649f;            // log(0.999*7/0.001), softmax_scale(8)
    const float kLog2e = 1.4426950408889634f;

    if (lane < 8) {
        float s = 1.0f;
        for (int v = 0; v < 4; ++v) {
            float po = 1.0f;
            const float* u = &uni[((b * 8 + lane) * 4 + v) * 25];
            for (int o = 0; o < 25; ++o) po *= (1.0f - u[o]);
            s *= (1.0f - po);
        }
        sv[lane] = s * kScale;
    }
    __syncthreads();
    if (lane == 0) {
        float mx = sv[0];
        for (int t = 1; t < 8; ++t) mx = fmaxf(mx, sv[t]);
        red[0] = mx;
    }
    __syncthreads();
    // exp(x) = 2^(x*log2e); argument <= 0, finite -> bare v_exp_f32 is safe
    if (lane < 8) sv[lane] = fast_exp2((sv[lane] - red[0]) * kLog2e);
    __syncthreads();
    if (lane == 0) {
        float sm = 0.0f;
        for (int t = 0; t < 8; ++t) sm += sv[t];
        red[1] = sm;
    }
    __syncthreads();
    if (lane < 8) out[51200 + b * 8 + lane] = sv[lane] / red[1];
}

// ---------------------------------------------------------------------------
extern "C" void kernel_launch(void* const* d_in, const int* in_sizes, int n_in,
                              void* d_out, int out_size, void* d_ws, size_t ws_size,
                              hipStream_t stream) {
    const float* iuf = (const float*)d_in[0];  // inv_unary_feats  (8,4,64)
    const float* ibf = (const float*)d_in[1];  // inv_binary_feats (8,4,3,16)
    const float* uf  = (const float*)d_in[2];  // unary_feats      (64,25,64)
    const float* bf  = (const float*)d_in[3];  // binary_feats     (64,25,24,16)
    const float* iup = (const float*)d_in[4];  // inv_unaryp       (8,4,64)
    const float* ibp = (const float*)d_in[5];  // inv_binaryp      (8,4,3,16)
    float* out = (float*)d_out;                // uni_sets (51200) | inv_select (512)
    float* ws  = (float*)d_ws;

    k_conds <<<14,   256, 0, stream>>>(iuf, ibf, iup, ibp, ws);
    k_t1    <<<64,   256, 0, stream>>>(uf, bf, ws);
    k_uni   <<<1600, 32,  0, stream>>>(bf, ws, out);
    k_select<<<64,   32,  0, stream>>>(out, out);
}
